// dqn_35493609734393
// MI455X (gfx1250) — matrix operand order repaired and checked by reading, not compiled, not run on hardware
//
#include <hip/hip_runtime.h>

// CDNA5 / gfx1250 WMMA MLP inference kernel, round 3.
//
// D(out x 16batch) = A(weights, bias-augmented) x B(activations + constant-1 row), C = inline 0.
// - Bias folded via augmented column; layers 0+1 fused (no ReLU between them): 5 WMMAs/tile.
// - ReLU applied AFTER f16 conversion: cvt_pk results are canonical, so
//   __builtin_elementwise_max(half2) lowers to bare v_pk_max_num_f16 (no canonicalize max).
// - Two independent tiles per iteration: the second chain fills the WMMA->VALU
//   hazard slots (replaces v_nops) and improves dual-issue pairing.
// - B operands are loop-carried; repacks write only elements 0..5. Stale data in
//   elements 6..8 multiplies zero A columns (K=6,7,16) -> no re-zeroing movs.

typedef __attribute__((ext_vector_type(16))) _Float16 v16h;
typedef __attribute__((ext_vector_type(2)))  _Float16 h2;
typedef __attribute__((ext_vector_type(8)))  float    v8f;
typedef __attribute__((ext_vector_type(4)))  float    v4f;

#define IN_DIM   8
#define HID      5
#define OUT_DIM  4
#define TILE_M   16
#define THREADS  256
#define BLOCKS   2048

__device__ __forceinline__ v16h zero16h() {
    v16h z;
#pragma unroll
    for (int i = 0; i < 16; ++i) z[i] = (_Float16)0.0f;
    return z;
}

// Weight A-matrix for a K=5 layer, bias augmented at K=5.
__device__ __forceinline__ v16h make_A5(const float* __restrict__ W,
                                        const float* __restrict__ b,
                                        int out_dim, int lane) {
    v16h a = zero16h();
    if (lane < out_dim) {
        const float* wr = W + lane * HID;
#pragma unroll
        for (int k = 0; k < HID; ++k) a[k] = (_Float16)wr[k];
        a[HID] = (_Float16)b[lane];          // bias column, K=5
    }
    return a;
}

// D -> next-layer B operand: cvt f32->f16 (packed), then ReLU in f16
// (v_pk_max_num_f16, no canonicalize). Constant-1 bias element at K=5 comes
// from cvt_pk(d4, 1.0) with an inline constant; relu(1)=1. Writes B[0..5] only.
__device__ __forceinline__ void repack_relu(v16h& B, v8f d) {
    h2 z2; z2[0] = (_Float16)0.0f; z2[1] = (_Float16)0.0f;
    h2 p0, p1, p2;
    p0[0] = (_Float16)d[0]; p0[1] = (_Float16)d[1];
    p1[0] = (_Float16)d[2]; p1[1] = (_Float16)d[3];
    p2[0] = (_Float16)d[4]; p2[1] = (_Float16)1.0f;
    p0 = __builtin_elementwise_max(p0, z2);
    p1 = __builtin_elementwise_max(p1, z2);
    p2 = __builtin_elementwise_max(p2, z2);
    B[0] = p0[0]; B[1] = p0[1];
    B[2] = p1[0]; B[3] = p1[1];
    B[4] = p2[0]; B[5] = p2[1];
}

// Input-layer B: x row in elements 0..7 (K=0..7). Element 8 (K=16, the bias-1)
// is set once outside the loop and never clobbered.
__device__ __forceinline__ void build_Bx(v16h& B, v4f x0, v4f x1) {
#pragma unroll
    for (int k = 0; k < 4; ++k) {
        B[k]     = (_Float16)x0[k];
        B[4 + k] = (_Float16)x1[k];
    }
}

__global__ __launch_bounds__(THREADS) void
dqn_mlp_wmma_kernel(const float* __restrict__ x,
                    const float* __restrict__ W0, const float* __restrict__ b0,
                    const float* __restrict__ W1, const float* __restrict__ b1,
                    const float* __restrict__ W2, const float* __restrict__ b2,
                    const float* __restrict__ W3, const float* __restrict__ b3,
                    const float* __restrict__ W4, const float* __restrict__ b4,
                    const float* __restrict__ W5, const float* __restrict__ b5,
                    float* __restrict__ out, int rows) {
    const int lane  = threadIdx.x & 31;
    const int gwave = (int)((blockIdx.x * THREADS + threadIdx.x) >> 5);
    const int nwave = (int)((gridDim.x * THREADS) >> 5);
    const int tiles = (rows + TILE_M - 1) / TILE_M;

    // ---- per-wave setup (amortized over the grid-stride loop) ----

    // Fused layer 0+1: Wf = W1*W0 (5x8), bf = W1*b0 + b1 (f32, cvt once).
    // Bias column K=16 lives in lanes 0-15, element 8 (16-bit A layout).
    v16h Af = zero16h();
    if (lane < HID) {
        const float* w1r = W1 + lane * HID;
#pragma unroll
        for (int k = 0; k < IN_DIM; ++k) {
            float s = 0.0f;
#pragma unroll
            for (int j = 0; j < HID; ++j) s += w1r[j] * W0[j * IN_DIM + k];
            Af[k] = (_Float16)s;
        }
    }
    if (lane < HID) {
        const int m = lane;
        const float* w1r = W1 + m * HID;
        float s = b1[m];
#pragma unroll
        for (int j = 0; j < HID; ++j) s += w1r[j] * b0[j];
        Af[8] = (_Float16)s;                 // bias column, K=16
    }

    const v16h A2 = make_A5(W2, b2, HID,     lane);
    const v16h A3 = make_A5(W3, b3, HID,     lane);
    const v16h A4 = make_A5(W4, b4, HID,     lane);
    const v16h A5 = make_A5(W5, b5, OUT_DIM, lane);

    v8f c0;                                  // zero accumulator -> inline 0 SRC2
#pragma unroll
    for (int r = 0; r < 8; ++r) c0[r] = 0.0f;

    v4f z4;
#pragma unroll
    for (int r = 0; r < 4; ++r) z4[r] = 0.0f;

    // Loop-carried B operands: elements 9..15 zero forever; element 8 = 1.0
    // (input-layer bias row; K=16 column of hidden A's is zero -> harmless later).
    v16h Ba = zero16h(); Ba[IN_DIM] = (_Float16)1.0f;
    v16h Bb = zero16h(); Bb[IN_DIM] = (_Float16)1.0f;

    // Lanes 16-31 duplicate rows 0-15 (their B elements are K=8..15,24..31 -> zero A cols).
    const int lrow = lane & 15;

    // Two adjacent tiles per iteration.
    for (int t = gwave * 2; t < tiles; t += nwave * 2) {
        const int row0a = t * TILE_M;
        const int row0b = row0a + TILE_M;

        int ra = row0a + lrow; if (ra >= rows) ra = rows - 1;
        int rb = row0b + lrow; if (rb >= rows) rb = rows - 1;
        const float* xa = x + (size_t)ra * IN_DIM;
        const float* xb = x + (size_t)rb * IN_DIM;
        v4f xa0 = *(const v4f*)(xa);
        v4f xa1 = *(const v4f*)(xa + 4);
        v4f xb0 = *(const v4f*)(xb);
        v4f xb1 = *(const v4f*)(xb + 4);

        build_Bx(Ba, xa0, xa1);
        build_Bx(Bb, xb0, xb1);

        v8f da = __builtin_amdgcn_wmma_f32_16x16x32_f16(false, Af, false, Ba,
                                                        (short)0, c0, false, false);
        v8f db = __builtin_amdgcn_wmma_f32_16x16x32_f16(false, Af, false, Bb,
                                                        (short)0, c0, false, false);
        repack_relu(Ba, da);
        repack_relu(Bb, db);

        da = __builtin_amdgcn_wmma_f32_16x16x32_f16(false, A2, false, Ba,
                                                    (short)0, c0, false, false);
        db = __builtin_amdgcn_wmma_f32_16x16x32_f16(false, A2, false, Bb,
                                                    (short)0, c0, false, false);
        repack_relu(Ba, da);
        repack_relu(Bb, db);

        da = __builtin_amdgcn_wmma_f32_16x16x32_f16(false, A3, false, Ba,
                                                    (short)0, c0, false, false);
        db = __builtin_amdgcn_wmma_f32_16x16x32_f16(false, A3, false, Bb,
                                                    (short)0, c0, false, false);
        repack_relu(Ba, da);
        repack_relu(Bb, db);

        da = __builtin_amdgcn_wmma_f32_16x16x32_f16(false, A4, false, Ba,
                                                    (short)0, c0, false, false);
        db = __builtin_amdgcn_wmma_f32_16x16x32_f16(false, A4, false, Bb,
                                                    (short)0, c0, false, false);
        repack_relu(Ba, da);
        repack_relu(Bb, db);

        da = __builtin_amdgcn_wmma_f32_16x16x32_f16(false, A5, false, Ba,
                                                    (short)0, c0, false, false);
        db = __builtin_amdgcn_wmma_f32_16x16x32_f16(false, A5, false, Bb,
                                                    (short)0, c0, false, false);

        // Output: lane n<16 holds out[0..3] for its batch row in d[0..3] (f32).
        if (lane < 16) {
            if (row0a + lane < rows) {
                v4f o;
#pragma unroll
                for (int q = 0; q < OUT_DIM; ++q) o[q] = da[q];
                o = __builtin_elementwise_max(o, z4);
                *(v4f*)(out + (size_t)(row0a + lane) * OUT_DIM) = o;
            }
            if (row0b + lane < rows) {
                v4f o;
#pragma unroll
                for (int q = 0; q < OUT_DIM; ++q) o[q] = db[q];
                o = __builtin_elementwise_max(o, z4);
                *(v4f*)(out + (size_t)(row0b + lane) * OUT_DIM) = o;
            }
        }
    }
}

extern "C" void kernel_launch(void* const* d_in, const int* in_sizes, int n_in,
                              void* d_out, int out_size, void* d_ws, size_t ws_size,
                              hipStream_t stream) {
    const float* x  = (const float*)d_in[0];
    const float* W0 = (const float*)d_in[1];
    const float* b0 = (const float*)d_in[2];
    const float* W1 = (const float*)d_in[3];
    const float* b1 = (const float*)d_in[4];
    const float* W2 = (const float*)d_in[5];
    const float* b2 = (const float*)d_in[6];
    const float* W3 = (const float*)d_in[7];
    const float* b3 = (const float*)d_in[8];
    const float* W4 = (const float*)d_in[9];
    const float* b4 = (const float*)d_in[10];
    const float* W5 = (const float*)d_in[11];
    const float* b5 = (const float*)d_in[12];
    float* out = (float*)d_out;

    const int rows = in_sizes[0] / IN_DIM;   // 2,097,152

    dqn_mlp_wmma_kernel<<<dim3(BLOCKS), dim3(THREADS), 0, stream>>>(
        x, W0, b0, W1, b1, W2, b2, W3, b3, W4, b4, W5, b5, out, rows);
}
